// GCN_AE__47296179863980
// MI455X (gfx1250) — matrix operand order repaired and checked by reading, not compiled, not run on hardware
//
#include <hip/hip_runtime.h>
#include <hip/hip_bf16.h>
#include <math.h>

// ---------------- CDNA5 WMMA plumbing ----------------
typedef __attribute__((ext_vector_type(16))) __bf16 v16bf;
typedef __attribute__((ext_vector_type(8)))  float  v8f;

__device__ __forceinline__ v8f wmma_bf16(v16bf a, v16bf b, v8f c) {
  // D = A(16x32 bf16) * B(32x16 bf16) + C(f32)
  return __builtin_amdgcn_wmma_f32_16x16x32_bf16(false, a, false, b, (short)0, c,
                                                 false, false);
}
__device__ __forceinline__ __bf16 f2bf(float x) { return (__bf16)x; }

// ---- async global->LDS copy (CDNA5, tracked by ASYNCcnt) ----
typedef __attribute__((address_space(3))) char lds_byte;
__device__ __forceinline__ void async_b32(void* lp, const void* g) {
  unsigned l = (unsigned)(unsigned long long)(lds_byte*)lp;   // LDS byte offset
  asm volatile("global_load_async_to_lds_b32 %0, %1, off"
               :: "v"(l), "v"((unsigned long long)g) : "memory");
}
__device__ __forceinline__ void wait_async0() {
  asm volatile("s_wait_asynccnt 0x0" ::: "memory");
}

#define B_  32
#define N_  1000
#define C_  9
#define E_  32000

#define FSTRIDE 36   // f32 elems; 144B row: 16B aligned, start banks l*36%64 all distinct

static inline int ceil_div(int a, int b) { return (a + b - 1) / b; }

// ---------------- utility kernels ----------------
__global__ void k_fill0(float* p, int n) {
  int t = blockIdx.x * blockDim.x + threadIdx.x;
  if (t < n) p[t] = 0.0f;
}

// eb[j][b*E+k] = e_[j][k] + b*N
__global__ void k_build_eb(const int* __restrict__ e, int* __restrict__ r,
                           int* __restrict__ c) {
  int t = blockIdx.x * blockDim.x + threadIdx.x;
  if (t >= B_ * E_) return;
  int b = t / E_, k = t - b * E_;
  r[t] = e[k] + b * N_;
  c[t] = e[E_ + k] + b * N_;
}

__global__ void k_deg(const int* __restrict__ row, int nE, float* __restrict__ deg) {
  int t = blockIdx.x * blockDim.x + threadIdx.x;
  if (t < nE) atomicAdd(&deg[row[t]], 1.0f);
}

__global__ void k_norm(const int* __restrict__ row, const int* __restrict__ col,
                       const float* __restrict__ deg, float* __restrict__ nrm, int nE) {
  int t = blockIdx.x * blockDim.x + threadIdx.x;
  if (t >= nE) return;
  float dr = deg[row[t]], dc = deg[col[t]];
  float a = dr > 0.0f ? rsqrtf(dr) : 0.0f;
  float b = dc > 0.0f ? rsqrtf(dc) : 0.0f;
  nrm[t] = a * b;
}

// out[col[e], ch] += norm[e] * h[row[e], ch]
__global__ void k_gcn_scatter(const float* __restrict__ h, const float* __restrict__ nrm,
                              const int* __restrict__ row, const int* __restrict__ col,
                              float* __restrict__ out, int cout, int nE) {
  long t = (long)blockIdx.x * blockDim.x + threadIdx.x;
  if (t >= (long)nE * cout) return;
  int e  = (int)(t / cout);
  int ch = (int)(t - (long)e * cout);
  atomicAdd(&out[(long)col[e] * cout + ch], nrm[e] * h[(long)row[e] * cout + ch]);
}

// AS[row[e], j] += S[col[e], j]
__global__ void k_as_scatter(const float* __restrict__ S, const int* __restrict__ row,
                             const int* __restrict__ col, float* __restrict__ AS,
                             int M, int nE) {
  long t = (long)blockIdx.x * blockDim.x + threadIdx.x;
  if (t >= (long)nE * M) return;
  int e = (int)(t / M);
  int j = (int)(t - (long)e * M);
  atomicAdd(&AS[(long)row[e] * M + j], S[(long)col[e] * M + j]);
}

__global__ void k_softmax(float* __restrict__ S, int rows, int M) {
  int r = blockIdx.x * blockDim.x + threadIdx.x;
  if (r >= rows) return;
  float* p = S + (long)r * M;
  float mx = -3.4e38f;
  for (int j = 0; j < M; ++j) mx = fmaxf(mx, p[j]);
  float s = 0.0f;
  for (int j = 0; j < M; ++j) { float e = expf(p[j] - mx); p[j] = e; s += e; }
  float inv = 1.0f / s;
  for (int j = 0; j < M; ++j) p[j] *= inv;
}

// Zp (first 3 channels of x_new) and cluster weight w
__global__ void k_zp_w(const float* __restrict__ S, const float* __restrict__ X, int ldx,
                       float* __restrict__ xnew, int cout, int Np, int M) {
  int t = blockIdx.x * blockDim.x + threadIdx.x;
  if (t >= B_ * M) return;
  int b = t / M, m = t - b * M;
  const float* Sb = S + (long)b * Np * M + m;
  const float* Xb = X + (long)b * Np * ldx;
  float w = 0.f, z0 = 0.f, z1 = 0.f, z2 = 0.f;
  for (int n = 0; n < Np; ++n) {
    float s = Sb[(long)n * M];
    const float* xp = Xb + (long)n * ldx;
    w += s; z0 += s * xp[0]; z1 += s * xp[1]; z2 += s * xp[2];
  }
  w = fmaxf(w, 1e-10f);
  float* o = xnew + (long)t * cout;
  o[0] = z0 / w; o[1] = z1 / w; o[2] = z2 / w;
}

// top-4 per row of A_MM -> new edge list (ties keep lowest index, matching lax.top_k)
__global__ void k_topk(const float* __restrict__ AMM, int M,
                       int* __restrict__ orow, int* __restrict__ ocol) {
  int t = blockIdx.x * blockDim.x + threadIdx.x;
  if (t >= B_ * M) return;
  int b = t / M, m = t - b * M;
  const float* r = AMM + (long)b * M * M + (long)m * M;
  float bv0 = -3.4e38f, bv1 = -3.4e38f, bv2 = -3.4e38f, bv3 = -3.4e38f;
  int bi0 = 0, bi1 = 0, bi2 = 0, bi3 = 0;
  for (int j = 0; j < M; ++j) {
    float v = r[j];
    if (v > bv0)      { bv3=bv2;bi3=bi2; bv2=bv1;bi2=bi1; bv1=bv0;bi1=bi0; bv0=v;bi0=j; }
    else if (v > bv1) { bv3=bv2;bi3=bi2; bv2=bv1;bi2=bi1; bv1=v; bi1=j; }
    else if (v > bv2) { bv3=bv2;bi3=bi2; bv2=v; bi2=j; }
    else if (v > bv3) { bv3=v; bi3=j; }
  }
  int base = t * 4, src = b * M + m, off = b * M;
  orow[base+0]=src; orow[base+1]=src; orow[base+2]=src; orow[base+3]=src;
  ocol[base+0]=off+bi0; ocol[base+1]=off+bi1; ocol[base+2]=off+bi2; ocol[base+3]=off+bi3;
}

// head output: channels 3..5 sigmoid, rest copied
__global__ void k_head_final(const float* __restrict__ h, float* __restrict__ out, int n) {
  int t = blockIdx.x * blockDim.x + threadIdx.x;
  if (t >= n) return;
  int ch = t % C_;
  float v = h[t];
  if (ch >= 3 && ch < 6) v = 1.0f / (1.0f + expf(-v));
  out[t] = v;
}

// ---------------- LDS staging (f32 tiles, async fast path) ----------------
// A tile (16 rows x 32 k, row-major src) -> Af[row*FSTRIDE + k]
__device__ __forceinline__ void stageA_row(const float* __restrict__ A, int lda,
                                           int m0, int k0, int Mr, int Kt,
                                           float* Af, int t) {
  int row = t >> 3;           // 0..15
  int kc  = (t & 7) * 4;      // 0,4,...,28
  if ((m0 + 16 <= Mr) && (k0 + 32 <= Kt)) {
    const float* src = A + (long)(m0 + row) * lda + k0 + kc;
#pragma unroll
    for (int i = 0; i < 4; ++i) async_b32(&Af[row * FSTRIDE + kc + i], src + i);
  } else {
    int rr = min(m0 + row, Mr - 1);
    bool rok = (m0 + row) < Mr;
#pragma unroll
    for (int i = 0; i < 4; ++i) {
      int k = k0 + kc + i, kk = min(k, Kt - 1);
      float v = A[(long)rr * lda + kk];        // clamped, always valid
      Af[row * FSTRIDE + kc + i] = (rok && k < Kt) ? v : 0.0f;
    }
  }
}

// A^T tile: src S[k, m] (k0..k0+31, m0..m0+15) -> Af[m*FSTRIDE + k]
__device__ __forceinline__ void stageA_T(const float* __restrict__ S, int lda,
                                         int m0, int k0, int Mr, int Kt,
                                         float* Af, int t) {
  int k  = t >> 2;            // 0..31
  int mc = (t & 3) * 4;       // 0,4,8,12
  if ((k0 + 32 <= Kt) && (m0 + 16 <= Mr)) {
    const float* src = S + (long)(k0 + k) * lda + m0 + mc;
#pragma unroll
    for (int i = 0; i < 4; ++i) async_b32(&Af[(mc + i) * FSTRIDE + k], src + i);
  } else {
    int kk = min(k0 + k, Kt - 1);
    bool kok = (k0 + k) < Kt;
    const float* src = S + (long)kk * lda;
#pragma unroll
    for (int i = 0; i < 4; ++i) {
      int m = m0 + mc + i, mm = min(m, Mr - 1);
      float v = src[mm];
      Af[(mc + i) * FSTRIDE + k] = (kok && m < Mr) ? v : 0.0f;
    }
  }
}

// B tile (32 k x 64 n, row-major src) transposed -> BTf[n*FSTRIDE + k]
__device__ __forceinline__ void stageB_T(const float* __restrict__ W, int ldw,
                                         int n0, int k0, int Nc, int Kt,
                                         float* BTf, int t) {
  int k   = t >> 2;           // 0..31
  int ncb = (t & 3) * 16;     // 0,16,32,48
  if ((k0 + 32 <= Kt) && (n0 + 64 <= Nc)) {
    const float* src = W + (long)(k0 + k) * ldw + n0 + ncb;
#pragma unroll
    for (int i = 0; i < 16; ++i) async_b32(&BTf[(ncb + i) * FSTRIDE + k], src + i);
  } else {
    int kk = min(k0 + k, Kt - 1);
    bool kok = (k0 + k) < Kt;
    const float* src = W + (long)kk * ldw;
#pragma unroll
    for (int i = 0; i < 16; ++i) {
      int n = n0 + ncb + i, nn = min(n, Nc - 1);
      float v = src[nn];
      BTf[(ncb + i) * FSTRIDE + k] = (kok && n < Nc) ? v : 0.0f;
    }
  }
}

// Fragment reads from f32 LDS (ds_load_b128 x2/x4) + cvt to bf16
__device__ __forceinline__ v16bf frag_a_f32(const float* Af, int lane) {
  int row = lane & 15;
  int kb  = (lane >> 4) << 4;                 // 0 or 16
  const float* p = Af + row * FSTRIDE + kb;
  v16bf a;
#pragma unroll
  for (int j = 0; j < 8; ++j) a[j] = f2bf(p[j]);
#pragma unroll
  for (int j = 0; j < 8; ++j) a[8 + j] = f2bf(p[8 + j]);
  return a;
}
__device__ __forceinline__ v16bf frag_b_f32(const float* BTf, int nloc, int lane) {
  int kb = (lane >> 4) << 4;                  // 0 or 16
  const float* p = BTf + nloc * FSTRIDE + kb;
  v16bf b;
#pragma unroll
  for (int j = 0; j < 16; ++j) b[j] = f2bf(p[j]);
  return b;
}

// ---------------- WMMA GEMM kernels ----------------
// Double-buffered: async-stage tile i+1 while WMMA consumes tile i.
__global__ void k_gemm_bias(const float* __restrict__ A, int lda,
                            const float* __restrict__ W, int ldw,
                            const float* __restrict__ bias,
                            float* __restrict__ Cp, int ldc,
                            int Mr, int Nc, int Kt) {
  __shared__ alignas(16) float Af[2][16 * FSTRIDE];
  __shared__ alignas(16) float BTf[2][64 * FSTRIDE];
  const int t = threadIdx.x;
  const int lane = t & 31, wv = t >> 5;
  const int n0 = blockIdx.x * 64;
  const int m0 = blockIdx.y * 16;
  const int nloc = wv * 16 + (lane & 15);
  const int steps = (Kt + 31) >> 5;
  v8f acc = {};
  stageA_row(A, lda, m0, 0, Mr, Kt, Af[0], t);
  stageB_T (W, ldw, n0, 0, Nc, Kt, BTf[0], t);
  int cur = 0;
  for (int i = 0; i < steps; ++i) {
    wait_async0();       // my async copies into buf[cur] (issued last iter) done
    __syncthreads();     // everyone's staging done; prior reads of buf[cur^1] done
    if (i + 1 < steps) {
      stageA_row(A, lda, m0, (i + 1) << 5, Mr, Kt, Af[cur ^ 1], t);
      stageB_T (W, ldw, n0, (i + 1) << 5, Nc, Kt, BTf[cur ^ 1], t);
    }
    v16bf a = frag_a_f32(Af[cur], lane);
    v16bf b = frag_b_f32(BTf[cur], nloc, lane);
    acc = wmma_bf16(a, b, acc);
    cur ^= 1;
  }
  int col = n0 + nloc;
  if (col < Nc) {
    float bv = bias ? bias[col] : 0.0f;
    int rbase = m0 + ((lane >> 4) << 3);
#pragma unroll
    for (int v = 0; v < 8; ++v) {
      int rr = rbase + v;
      if (rr < Mr) Cp[(long)rr * ldc + col] = acc[v] + bv;
    }
  }
}

// Batched transposed GEMM: C[b][m,n] = sum_k A[b][k,m] * Bm[b][k,n]
__global__ void k_gemm_at_b(const float* __restrict__ A, int lda, long strideA,
                            const float* __restrict__ Bm, int ldb, long strideB,
                            float* __restrict__ Cp, int ldc, long strideC,
                            int Mr, int Nc, int Kt) {
  A  += (long)blockIdx.z * strideA;
  Bm += (long)blockIdx.z * strideB;
  Cp += (long)blockIdx.z * strideC;
  __shared__ alignas(16) float Af[2][16 * FSTRIDE];
  __shared__ alignas(16) float BTf[2][64 * FSTRIDE];
  const int t = threadIdx.x;
  const int lane = t & 31, wv = t >> 5;
  const int n0 = blockIdx.x * 64;
  const int m0 = blockIdx.y * 16;
  const int nloc = wv * 16 + (lane & 15);
  const int steps = (Kt + 31) >> 5;
  v8f acc = {};
  stageA_T(A, lda, m0, 0, Mr, Kt, Af[0], t);
  stageB_T(Bm, ldb, n0, 0, Nc, Kt, BTf[0], t);
  int cur = 0;
  for (int i = 0; i < steps; ++i) {
    wait_async0();
    __syncthreads();
    if (i + 1 < steps) {
      stageA_T(A, lda, m0, (i + 1) << 5, Mr, Kt, Af[cur ^ 1], t);
      stageB_T(Bm, ldb, n0, (i + 1) << 5, Nc, Kt, BTf[cur ^ 1], t);
    }
    v16bf a = frag_a_f32(Af[cur], lane);
    v16bf b = frag_b_f32(BTf[cur], nloc, lane);
    acc = wmma_bf16(a, b, acc);
    cur ^= 1;
  }
  int col = n0 + nloc;
  if (col < Nc) {
    int rbase = m0 + ((lane >> 4) << 3);
#pragma unroll
    for (int v = 0; v < 8; ++v) {
      int rr = rbase + v;
      if (rr < Mr) Cp[(long)rr * ldc + col] = acc[v];
    }
  }
}

// ---------------- host-side layer runners ----------------
static void gemm_bias(const float* A, int lda, const float* W, int ldw,
                      const float* bias, float* Cp, int ldc,
                      int Mr, int Nc, int Kt, hipStream_t s) {
  dim3 g(ceil_div(Nc, 64), ceil_div(Mr, 16));
  k_gemm_bias<<<g, 128, 0, s>>>(A, lda, W, ldw, bias, Cp, ldc, Mr, Nc, Kt);
}

static void run_gcn(const float* X, int BN, int cin, int cout,
                    const float* W, const float* bW, const float* Wr, const float* br,
                    const int* erow, const int* ecol, int nE,
                    float* out, float* H, float* deg, float* nrm, hipStream_t s) {
  gemm_bias(X, cin, Wr, cout, br, out, cout, BN, cout, cin, s);  // y
  gemm_bias(X, cin, W,  cout, bW, H,   cout, BN, cout, cin, s);  // h
  k_fill0<<<ceil_div(BN, 256), 256, 0, s>>>(deg, BN);
  k_deg<<<ceil_div(nE, 256), 256, 0, s>>>(erow, nE, deg);
  k_norm<<<ceil_div(nE, 256), 256, 0, s>>>(erow, ecol, deg, nrm, nE);
  long tot = (long)nE * cout;
  k_gcn_scatter<<<(unsigned)((tot + 255) / 256), 256, 0, s>>>(H, nrm, erow, ecol,
                                                              out, cout, nE);
}

static void run_pool(const float* X, int Np, int cin, int cout, int M,
                     const float* Wa, const float* ba, const float* We, const float* be,
                     const int* erow, const int* ecol, int nE,
                     float* xnew, int* orow, int* ocol,
                     float* S, float* F, float* AS, float* AMM, hipStream_t s) {
  const int BN = B_ * Np;
  const int cf = cout - 3;
  gemm_bias(X, cin, We, cf, be, F, cf, BN, cf, cin, s);       // f
  gemm_bias(X, cin, Wa, M,  ba, S, M,  BN, M,  cin, s);       // logits
  k_softmax<<<ceil_div(BN, 128), 128, 0, s>>>(S, BN, M);
  k_zp_w<<<ceil_div(B_ * M, 128), 128, 0, s>>>(S, X, cin, xnew, cout, Np, M);
  {   // Zf = S^T f  -> xnew[:, 3:]
    dim3 g(ceil_div(cf, 64), ceil_div(M, 16), B_);
    k_gemm_at_b<<<g, 128, 0, s>>>(S, M, (long)Np * M, F, cf, (long)Np * cf,
                                  xnew + 3, cout, (long)M * cout, M, cf, Np);
  }
  if (orow) {  // coarsened adjacency + top-k edges
    k_fill0<<<ceil_div(BN * M, 256), 256, 0, s>>>(AS, BN * M);
    long tot = (long)nE * M;
    k_as_scatter<<<(unsigned)((tot + 255) / 256), 256, 0, s>>>(S, erow, ecol, AS, M, nE);
    dim3 g(ceil_div(M, 64), ceil_div(M, 16), B_);
    k_gemm_at_b<<<g, 128, 0, s>>>(S, M, (long)Np * M, AS, M, (long)Np * M,
                                  AMM, M, (long)M * M, M, M, Np);
    k_topk<<<ceil_div(B_ * M, 128), 128, 0, s>>>(AMM, M, orow, ocol);
  }
}

// ---------------- entry point ----------------
extern "C" void kernel_launch(void* const* d_in, const int* in_sizes, int n_in,
                              void* d_out, int out_size, void* d_ws, size_t ws_size,
                              hipStream_t stream) {
  (void)in_sizes; (void)n_in; (void)out_size; (void)ws_size;
  const float* x  = (const float*)d_in[0];
  const int*   eI = (const int*)d_in[1];
  const float* P[52];
  for (int i = 0; i < 52; ++i) P[i] = (const float*)d_in[i];

  // workspace layout (256B aligned slabs)
  char* w = (char*)d_ws;
  size_t off = 0;
  auto allocF = [&](size_t n) { float* p = (float*)(w + off); off += ((n * 4 + 255) / 256) * 256; return p; };
  auto allocI = [&](size_t n) { int*   p = (int*)  (w + off); off += ((n * 4 + 255) / 256) * 256; return p; };

  float* XA  = allocF(2048000);   // <= 32000 x 64
  float* XB  = allocF(2048000);
  float* H   = allocF(2048000);   // h / f scratch
  float* S   = allocF(8192000);   // <= 32000 x 256 (and 8192 x 1000)
  float* AS  = allocF(8192000);
  float* AMM = allocF(2097152);   // 32 x 256 x 256
  float* deg = allocF(32000);
  float* nrm = allocF(1024000);
  int* E0r = allocI(1024000); int* E0c = allocI(1024000);
  int* E1r = allocI(32768);   int* E1c = allocI(32768);

  hipStream_t s = stream;
  float* zOut = (float*)d_out + B_ * N_ * C_;   // second output slab: z (512 x 256)

  // batched edge list
  k_build_eb<<<ceil_div(B_ * E_, 256), 256, 0, s>>>(eI, E0r, E0c);
  const int nE0 = B_ * E_;                                        // 1,024,000

  // ---- encoder ----
  run_gcn (x,   B_*N_,  C_,  64,  P[2],P[3],P[4],P[5],   E0r,E0c,nE0,   XA, H, deg, nrm, s);
  run_pool(XA,  N_,     64,  64,  256, P[6],P[7],P[8],P[9],  E0r,E0c,nE0,  XB, E1r,E1c, S,H,AS,AMM, s);
  run_gcn (XB,  B_*256, 64,  128, P[10],P[11],P[12],P[13], E1r,E1c,32768, XA, H, deg, nrm, s);
  run_pool(XA,  256,    128, 128, 64,  P[14],P[15],P[16],P[17], E1r,E1c,32768, XB, E0r,E0c, S,H,AS,AMM, s);
  run_gcn (XB,  B_*64,  128, 256, P[18],P[19],P[20],P[21], E0r,E0c,8192,  XA, H, deg, nrm, s);
  run_pool(XA,  64,     256, 256, 16,  P[22],P[23],P[24],P[25], E0r,E0c,8192, zOut, E1r,E1c, S,H,AS,AMM, s);

  // ---- decoder ----
  run_gcn (zOut, B_*16, 256, 128, P[26],P[27],P[28],P[29], E1r,E1c,2048,  XB, H, deg, nrm, s);
  run_pool(XB,  16,     128, 128, 64,  P[30],P[31],P[32],P[33], E1r,E1c,2048, XA, E0r,E0c, S,H,AS,AMM, s);
  run_gcn (XA,  B_*64,  128, 64,  P[34],P[35],P[36],P[37], E0r,E0c,8192,  XB, H, deg, nrm, s);
  run_pool(XB,  64,     64,  64,  256, P[38],P[39],P[40],P[41], E0r,E0c,8192, XA, E1r,E1c, S,H,AS,AMM, s);
  run_gcn (XA,  B_*256, 64,  C_,  P[42],P[43],P[44],P[45], E1r,E1c,32768, XB, H, deg, nrm, s);
  // final pool: its e_new is never consumed -> skip AS / A_MM / top-k
  run_pool(XB,  256,    C_,  C_,  1000, P[46],P[47],P[48],P[49], nullptr,nullptr,0,
           XA, nullptr,nullptr, S,H,AS,AMM, s);

  // ---- head + output mix ----
  gemm_bias(XA, C_, P[50], C_, P[51], XB, C_, B_ * N_, C_, C_, s);
  k_head_final<<<ceil_div(B_ * N_ * C_, 256), 256, 0, s>>>(XB, (float*)d_out, B_ * N_ * C_);
}
